// BayesianMultiHeadAttention_62010737819701
// MI455X (gfx1250) — matrix operand order repaired and checked by reading, not compiled, not run on hardware
//
#include <hip/hip_runtime.h>
#include <hip/hip_bf16.h>

#define D_MODEL 1024
#define NHEADS  16
#define DK      64
#define BATCH   8
#define SEQ     1024
#define MROWS   (BATCH*SEQ)   /* 8192 */
#define DRAW    16
#define SM_SCALE 0.125f       /* 1/sqrt(64) */

typedef __bf16 bf16;
typedef __attribute__((ext_vector_type(16))) __bf16 v16bf;
typedef __attribute__((ext_vector_type(8)))  float  v8f;

union V16 { v16bf v; unsigned u[8]; unsigned short h[16]; };

__device__ __forceinline__ unsigned short bfbits(float f) {
  unsigned u = __float_as_uint(f);
  return (unsigned short)((u + 0x7FFFu + ((u >> 16) & 1u)) >> 16);
}
__device__ __forceinline__ bf16 f2bf(float f) {
  return __builtin_bit_cast(bf16, bfbits(f));
}
__device__ __forceinline__ float bfu2f(unsigned short s) {
  return __uint_as_float(((unsigned)s) << 16);
}
__device__ __forceinline__ float swapx16(float x) {   // exchange with lane^16 (SWAPX16)
  return __int_as_float(__builtin_amdgcn_ds_swizzle(__float_as_int(x), 0x401f));
}
__device__ __forceinline__ unsigned swapx16u(unsigned x) {
  return (unsigned)__builtin_amdgcn_ds_swizzle((int)x, 0x401f);
}
__device__ __forceinline__ unsigned pack2bf(float lo, float hi) {
  return (unsigned)bfbits(lo) | ((unsigned)bfbits(hi) << 16);
}
__device__ __forceinline__ v8f wmma_bf16(v16bf a, v16bf b, v8f c) {
  return __builtin_amdgcn_wmma_f32_16x16x32_bf16(false, a, false, b, (short)0, c, false, false);
}
// A-operand loader (16-bit 16x32): elems 0..15 = K[hi*16..hi*16+15]
__device__ __forceinline__ v16bf load_a(const bf16* rowbase, int hi) {
  V16 r;
  *(uint4*)&r.u[0] = *(const uint4*)(rowbase + hi * 16);
  *(uint4*)&r.u[4] = *(const uint4*)(rowbase + hi * 16 + 8);
  return r.v;
}
// A-operand with op: 0 f32, 1 f32 squared, 2 bf16, 3 bf16 squared
template <int A_MODE>
__device__ __forceinline__ v16bf load_a_op(const void* Aptr, size_t rowIdx, int K, int k0, int hi) {
  V16 a;
  if constexpr (A_MODE <= 1) {
    const float* ap = (const float*)Aptr + rowIdx * (size_t)K + k0;
#pragma unroll
    for (int t = 0; t < 4; ++t) {
      float v0 = ap[hi * 16 + 2 * t],      v1 = ap[hi * 16 + 2 * t + 1];
      float v2 = ap[hi * 16 + 8 + 2 * t], v3 = ap[hi * 16 + 8 + 2 * t + 1];
      if constexpr (A_MODE == 1) { v0 *= v0; v1 *= v1; v2 *= v2; v3 *= v3; }
      a.u[t]     = pack2bf(v0, v1);
      a.u[4 + t] = pack2bf(v2, v3);
    }
  } else {
    const bf16* ap = (const bf16*)Aptr + rowIdx * (size_t)K + k0;
    *(uint4*)&a.u[0] = *(const uint4*)(ap + hi * 16);
    *(uint4*)&a.u[4] = *(const uint4*)(ap + hi * 16 + 8);
    if constexpr (A_MODE == 3) {
#pragma unroll
      for (int t = 0; t < 8; ++t) {
        float lo = bfu2f((unsigned short)(a.u[t] & 0xffffu));
        float hi2 = bfu2f((unsigned short)(a.u[t] >> 16));
        a.u[t] = pack2bf(lo * lo, hi2 * hi2);
      }
    }
  }
  return a.v;
}

// ---------------------------------------------------------------- prep kernels
// mode 0: bf16(src)    mode 1: bf16(softplus(src)^2)
__global__ void cvt_w_kernel(const float* __restrict__ src, bf16* __restrict__ dst,
                             int n, int mode) {
  int i = blockIdx.x * blockDim.x + threadIdx.x;
  if (i >= n) return;
  float v = src[i];
  if (mode == 1) { float sp = log1pf(__expf(v)); v = sp * sp; }
  dst[i] = f2bf(v);
}

__global__ void prep_x_kernel(const float* __restrict__ xr, bf16* __restrict__ xbf,
                              float* __restrict__ x2) {
  int i = blockIdx.x * blockDim.x + threadIdx.x;   // 0..MROWS-1
  if (i >= MROWS) return;
  const float* p = xr + (size_t)i * DRAW;
  float s = 0.f;
#pragma unroll
  for (int k = 0; k < DRAW; ++k) { float v = p[k]; s += v * v; xbf[(size_t)i * DRAW + k] = f2bf(v); }
  x2[i] = s;
}

// ---------------------------------------------------------------- WMMA GEMM
// C[M x N] = op(A)[M x K] @ W[N x K]^T + bias
// Each wave: 32(M) x 64(N) tile = 2 A-tiles x 4 B-tiles = 8 WMMAs per K-step.
// BIAS_MODE: 0 raw, 1 softplus(bias)^2
// OUT_MODE:  0 bf16 [M,N]; 1 bf16 per-head transposed Vt[b,h,d,n]; 2 f32 (+addbuf)
template <int A_MODE, int BIAS_MODE, int OUT_MODE>
__global__ __launch_bounds__(128)
void gemm_wmma_kernel(const void* __restrict__ Aptr, const bf16* __restrict__ W,
                      const float* __restrict__ bias, void* __restrict__ Cptr,
                      const float* __restrict__ addbuf, int M, int N, int K) {
  const int lane = threadIdx.x & 31;
  const int wave = threadIdx.x >> 5;
  const int row  = lane & 15;
  const int hi   = lane >> 4;
  const int m0 = (blockIdx.x * 4 + wave) * 32;
  const int n0 = blockIdx.y * 64;

  v8f acc[2][4] = {};

  const size_t arow0 = (size_t)(m0 + row);
  const size_t arow1 = (size_t)(m0 + 16 + row);
  const bf16* wrow = W + (size_t)(n0 + row) * K;

  for (int k0 = 0; k0 < K; k0 += 32) {
    v16bf a0 = load_a_op<A_MODE>(Aptr, arow0, K, k0, hi);
    v16bf a1 = load_a_op<A_MODE>(Aptr, arow1, K, k0, hi);
    const bf16* wp = wrow + k0 + hi * 16;
    v16bf b0 = *(const v16bf*)(wp);
    v16bf b1 = *(const v16bf*)(wp + (size_t)16 * K);
    v16bf b2 = *(const v16bf*)(wp + (size_t)32 * K);
    v16bf b3 = *(const v16bf*)(wp + (size_t)48 * K);
    // prefetch the streaming operands ~8 K-steps ahead (global_prefetch_b8)
    __builtin_prefetch(wp + 256, 0, 0);
    __builtin_prefetch((const char*)Aptr + (arow0 * (size_t)K + k0) * (A_MODE <= 1 ? 4 : 2) + 1024, 0, 0);
    acc[0][0] = wmma_bf16(a0, b0, acc[0][0]);
    acc[0][1] = wmma_bf16(a0, b1, acc[0][1]);
    acc[0][2] = wmma_bf16(a0, b2, acc[0][2]);
    acc[0][3] = wmma_bf16(a0, b3, acc[0][3]);
    acc[1][0] = wmma_bf16(a1, b0, acc[1][0]);
    acc[1][1] = wmma_bf16(a1, b1, acc[1][1]);
    acc[1][2] = wmma_bf16(a1, b2, acc[1][2]);
    acc[1][3] = wmma_bf16(a1, b3, acc[1][3]);
  }

#pragma unroll
  for (int i = 0; i < 2; ++i) {
#pragma unroll
    for (int r = 0; r < 8; ++r) {
      int m = m0 + i * 16 + r + hi * 8;
#pragma unroll
      for (int j = 0; j < 4; ++j) {
        int n = n0 + j * 16 + row;
        float val = acc[i][j][r];
        float bv = bias[n];
        if constexpr (BIAS_MODE == 1) { float sp = log1pf(__expf(bv)); bv = sp * sp; }
        val += bv;
        if constexpr (OUT_MODE == 0) {
          ((bf16*)Cptr)[(size_t)m * N + n] = f2bf(val);
        } else if constexpr (OUT_MODE == 1) {
          // n = h*64+d ; m = b*1024+seq ; Vt[b][h][d][seq]
          size_t idx = (size_t)(m >> 10) * (1u << 20) + (size_t)n * 1024 + (size_t)(m & 1023);
          ((bf16*)Cptr)[idx] = f2bf(val);
        } else {
          size_t idx = (size_t)m * N + n;
          float add = addbuf ? addbuf[idx] : 0.f;
          ((float*)Cptr)[idx] = val + add;
        }
      }
    }
  }
}

// ---------------------------------------------------------------- fused flash attention
// S^T = K_tile * Q^T (+ RBF via zero-padded bf16 wmma on x_raw), online softmax,
// ctx^T += V^T * P^T, var^T += Vv^T * (P*P)^T.  One wave handles (b,h) x 16 queries.
__global__ __launch_bounds__(128, 1)
void attn_flash_kernel(const bf16* __restrict__ Qb, const bf16* __restrict__ Kb,
                       const bf16* __restrict__ Vt, const bf16* __restrict__ Vvt,
                       const bf16* __restrict__ xbf, const float* __restrict__ x2,
                       const float* __restrict__ log_sigma_f,
                       const float* __restrict__ log_length,
                       bf16* __restrict__ ctx, float* __restrict__ var_attn) {
  const int lane = threadIdx.x & 31;
  const int wave = threadIdx.x >> 5;
  const int row = lane & 15, hi = lane >> 4;
  const int b = blockIdx.x >> 4, h = blockIdx.x & 15;
  const int q0 = (blockIdx.y * 4 + wave) * 16;
  const int q = q0 + row;

  const float sigma2 = __expf(2.f * log_sigma_f[h]);
  const float inv2l2 = 0.5f * __expf(-2.f * log_length[h]);
  const float x2q = x2[b * SEQ + q];

  // Q^T B-operands (lane = query col; 16 contiguous d at hi*16 within each 32-chunk)
  const bf16* qbase = Qb + ((size_t)(b * SEQ + q) * D_MODEL + h * DK + hi * 16);
  v16bf bq0 = *(const v16bf*)(qbase);
  v16bf bq1 = *(const v16bf*)(qbase + 32);
  // x_q B-operand: real d 0..15 on lanes<16, zero pad d 16..31 on lanes>=16
  v16bf bx = {};
  if (hi == 0) bx = *(const v16bf*)(xbf + (size_t)(b * SEQ + q) * DRAW);

  float m_run = -3.0e38f, l_run = 0.f;
  v8f cacc[4] = {};
  v8f wacc[4] = {};

  const bf16* kb_base = Kb  + (size_t)(b * SEQ) * D_MODEL + h * DK;
  const bf16* vb_base = Vt  + (size_t)((b * NHEADS + h) * DK + row) * SEQ;
  const bf16* vv_base = Vvt + (size_t)((b * NHEADS + h) * DK + row) * SEQ;
  const bf16* xk_base = xbf + (size_t)(b * SEQ) * DRAW;
  const float* x2_base = x2 + b * SEQ;

#pragma unroll 1
  for (int k0 = 0; k0 < SEQ; k0 += 32) {
    float st[2][8];
#pragma unroll
    for (int t = 0; t < 2; ++t) {
      int kb = k0 + t * 16;
      const bf16* kbase = kb_base + (size_t)(kb + row) * D_MODEL;
      v16bf a0 = load_a(kbase, hi);
      v16bf a1 = load_a(kbase + 32, hi);
      v8f s = {};
      s = wmma_bf16(a0, bq0, s);
      s = wmma_bf16(a1, bq1, s);
      // RBF cross term: A = x_keys (K padded to 32 with zeros), B = bx
      V16 ax; ax.v = v16bf{};
      if (hi == 0) ax.v = *(const v16bf*)(xk_base + (size_t)(kb + row) * DRAW);
      v8f g = {};
      g = wmma_bf16(ax.v, bx, g);
#pragma unroll
      for (int r = 0; r < 8; ++r) {
        float x2k = x2_base[kb + hi * 8 + r];
        float d2 = fmaxf(x2k + x2q - 2.f * g[r], 0.f);
        st[t][r] = s[r] * SM_SCALE + sigma2 * __expf(-d2 * inv2l2);
      }
    }
    // online softmax state (per query = per lane; lane^16 holds the other 16 keys)
    float tmax = st[0][0];
#pragma unroll
    for (int t = 0; t < 2; ++t)
#pragma unroll
      for (int r = 0; r < 8; ++r) tmax = fmaxf(tmax, st[t][r]);
    tmax = fmaxf(tmax, swapx16(tmax));
    float m_new = fmaxf(m_run, tmax);
    float scl = __expf(m_run - m_new);
    float scl2 = scl * scl;
    m_run = m_new;
    l_run *= scl;
#pragma unroll
    for (int dt = 0; dt < 4; ++dt)
#pragma unroll
      for (int r = 0; r < 8; ++r) { cacc[dt][r] *= scl; wacc[dt][r] *= scl2; }
    // probabilities overwrite the score tile in place
    float tsum = 0.f;
#pragma unroll
    for (int t = 0; t < 2; ++t)
#pragma unroll
      for (int r = 0; r < 8; ++r) { float e = __expf(st[t][r] - m_new); st[t][r] = e; tsum += e; }
    l_run += tsum + swapx16(tsum);

    // Build P / P^2 B-operands (K=32 keys, N=16 queries):
    // lane<16 needs keys 0..15 : own tile0 (0..7) + partner tile0 (8..15)
    // lane>=16 needs keys 16..31: partner tile1 (16..23) + own tile1 (24..31)
    V16 bp, bp2;
#pragma unroll
    for (int i = 0; i < 4; ++i) {
      unsigned u1 = pack2bf(st[0][2 * i], st[0][2 * i + 1]);
      unsigned u2 = pack2bf(st[1][2 * i], st[1][2 * i + 1]);
      unsigned s1 = pack2bf(st[0][2 * i] * st[0][2 * i], st[0][2 * i + 1] * st[0][2 * i + 1]);
      unsigned s2 = pack2bf(st[1][2 * i] * st[1][2 * i], st[1][2 * i + 1] * st[1][2 * i + 1]);
      unsigned o1 = swapx16u(u1), o2 = swapx16u(u2);
      unsigned t1 = swapx16u(s1), t2 = swapx16u(s2);
      bp.u[i]      = hi ? o2 : u1;
      bp.u[4 + i]  = hi ? u2 : o1;
      bp2.u[i]     = hi ? t2 : s1;
      bp2.u[4 + i] = hi ? s2 : t1;
    }
    // ctx^T += V^T * P^T ; var^T += Vv^T * (P^2)^T   (A rows = d, K = keys)
#pragma unroll
    for (int dt = 0; dt < 4; ++dt)
      cacc[dt] = wmma_bf16(load_a(vb_base + (size_t)dt * SEQ * 16 + k0, hi), bp.v, cacc[dt]);
#pragma unroll
    for (int dt = 0; dt < 4; ++dt)
      wacc[dt] = wmma_bf16(load_a(vv_base + (size_t)dt * SEQ * 16 + k0, hi), bp2.v, wacc[dt]);
  }

  float invl = 1.f / l_run, invl2 = invl * invl;
#pragma unroll
  for (int dt = 0; dt < 4; ++dt) {
#pragma unroll
    for (int r = 0; r < 8; ++r) {
      int d = dt * 16 + r + hi * 8;
      float cv = cacc[dt][r] * invl;
      float wv = wacc[dt][r] * invl2;
      size_t idx = (size_t)(b * SEQ + q) * D_MODEL + h * DK + d;
      ctx[idx] = f2bf(cv);
      var_attn[idx] = wv;
    }
  }
}

// ---------------------------------------------------------------- host
extern "C" void kernel_launch(void* const* d_in, const int* in_sizes, int n_in,
                              void* d_out, int out_size, void* d_ws, size_t ws_size,
                              hipStream_t stream) {
  (void)in_sizes; (void)n_in; (void)out_size; (void)ws_size;
  const float* H      = (const float*)d_in[0];
  const float* x_raw  = (const float*)d_in[1];
  const float* Wq_mu  = (const float*)d_in[2];
  const float* bq_mu  = (const float*)d_in[4];
  const float* Wk_mu  = (const float*)d_in[6];
  const float* bk_mu  = (const float*)d_in[8];
  const float* Wv_mu  = (const float*)d_in[10];
  const float* Wv_rho = (const float*)d_in[11];
  const float* bv_mu  = (const float*)d_in[12];
  const float* bv_rho = (const float*)d_in[13];
  const float* Wo_mu  = (const float*)d_in[14];
  const float* Wo_rho = (const float*)d_in[15];
  const float* bo_mu  = (const float*)d_in[16];
  const float* bo_rho = (const float*)d_in[17];
  const float* lsf    = (const float*)d_in[18];
  const float* llen   = (const float*)d_in[19];

  char* cur = (char*)d_ws;
  auto take = [&](size_t bytes) { void* p = cur; cur += (bytes + 255) & ~(size_t)255; return p; };
  const size_t ACT_BF = (size_t)MROWS * D_MODEL * sizeof(bf16);
  const size_t W_BF   = (size_t)D_MODEL * D_MODEL * sizeof(bf16);
  bf16*  Qbf      = (bf16*)take(ACT_BF);
  bf16*  Kbf      = (bf16*)take(ACT_BF);
  bf16*  Vtbf     = (bf16*)take(ACT_BF);          // [b,h,d,n]
  bf16*  Vvtbf    = (bf16*)take(ACT_BF);          // [b,h,d,n]
  bf16*  ctxbf    = (bf16*)take(ACT_BF);
  float* var_attn = (float*)take((size_t)MROWS * D_MODEL * sizeof(float));
  bf16*  Wqbf     = (bf16*)take(W_BF);
  bf16*  Wkbf     = (bf16*)take(W_BF);
  bf16*  Wvbf     = (bf16*)take(W_BF);
  bf16*  Wvvbf    = (bf16*)take(W_BF);
  bf16*  Wobf     = (bf16*)take(W_BF);
  bf16*  Wovbf    = (bf16*)take(W_BF);
  bf16*  xbf      = (bf16*)take((size_t)MROWS * DRAW * sizeof(bf16));
  float* x2       = (float*)take((size_t)MROWS * sizeof(float));

  const int wn = D_MODEL * D_MODEL;
  dim3 cb(256), cg((wn + 255) / 256);
  cvt_w_kernel<<<cg, cb, 0, stream>>>(Wq_mu,  Wqbf,  wn, 0);
  cvt_w_kernel<<<cg, cb, 0, stream>>>(Wk_mu,  Wkbf,  wn, 0);
  cvt_w_kernel<<<cg, cb, 0, stream>>>(Wv_mu,  Wvbf,  wn, 0);
  cvt_w_kernel<<<cg, cb, 0, stream>>>(Wv_rho, Wvvbf, wn, 1);
  cvt_w_kernel<<<cg, cb, 0, stream>>>(Wo_mu,  Wobf,  wn, 0);
  cvt_w_kernel<<<cg, cb, 0, stream>>>(Wo_rho, Wovbf, wn, 1);
  prep_x_kernel<<<(MROWS + 255) / 256, 256, 0, stream>>>(x_raw, xbf, x2);

  dim3 gg(MROWS / 128, D_MODEL / 64);   // wave = 32x64 tile, 4 waves/block
  // Q, K (normal layout), V and V_var (per-head transposed)
  gemm_wmma_kernel<0, 0, 0><<<gg, 128, 0, stream>>>(H, Wqbf,  bq_mu,  Qbf,   nullptr, MROWS, D_MODEL, D_MODEL);
  gemm_wmma_kernel<0, 0, 0><<<gg, 128, 0, stream>>>(H, Wkbf,  bk_mu,  Kbf,   nullptr, MROWS, D_MODEL, D_MODEL);
  gemm_wmma_kernel<0, 0, 1><<<gg, 128, 0, stream>>>(H, Wvbf,  bv_mu,  Vtbf,  nullptr, MROWS, D_MODEL, D_MODEL);
  gemm_wmma_kernel<1, 1, 1><<<gg, 128, 0, stream>>>(H, Wvvbf, bv_rho, Vvtbf, nullptr, MROWS, D_MODEL, D_MODEL);

  attn_flash_kernel<<<dim3(BATCH * NHEADS, SEQ / 64), 128, 0, stream>>>(
      Qbf, Kbf, Vtbf, Vvtbf, xbf, x2, lsf, llen, ctxbf, var_attn);

  float* out_mean = (float*)d_out;
  float* out_var  = out_mean + (size_t)MROWS * D_MODEL;
  gemm_wmma_kernel<2, 0, 2><<<gg, 128, 0, stream>>>(ctxbf, Wobf,  bo_mu,  out_mean, nullptr,  MROWS, D_MODEL, D_MODEL);
  gemm_wmma_kernel<3, 1, 2><<<gg, 128, 0, stream>>>(ctxbf, Wovbf, bo_rho, out_var,  var_attn, MROWS, D_MODEL, D_MODEL);
}
